// FGIN_71829033058360
// MI455X (gfx1250) — compile-verified, matrix operand order checked by reading, not run on hardware
//
#include <hip/hip_runtime.h>
#include <hip/hip_bf16.h>

// ---------------------------------------------------------------------------
// Types for CDNA5 WMMA
// ---------------------------------------------------------------------------
typedef __attribute__((ext_vector_type(16))) __bf16 v16bf;
typedef __attribute__((ext_vector_type(8)))  float  v8f;

union Frag16 { v16bf v; uint4 q[2]; };

__device__ inline unsigned short f2bf(float f) {
  unsigned u = __float_as_uint(f);
  unsigned r = u + 0x7FFFu + ((u >> 16) & 1u);   // round-to-nearest-even
  return (unsigned short)(r >> 16);
}

// ---------------------------------------------------------------------------
// Generic GEMM:  C[:, c_col0 : c_col0+Mo] = act( A[MxK] @ W[KxMo] + bias (+ addend) )
// A, W fp32 in global; converted to bf16 tiles in LDS; v_wmma_f32_16x16x32_bf16.
// Tile: BM=128, BN=64, BK=32. 256 threads = 8 waves, each wave 32x32 (2x2 WMMA).
// ---------------------------------------------------------------------------
#define BM 128
#define BN 64
#define BK 32
#define LDA_S 40   // LDS row stride (bf16 elems) for A tile  (80B, 16B aligned)
#define LDB_S 40   // LDS row stride for transposed W tile

__launch_bounds__(256)
__global__ void gemm_bf16_wmma(const float* __restrict__ A, int lda,
                               const float* __restrict__ W, int ldw,
                               const float* __restrict__ bias,
                               const float* __restrict__ addend, int ld_add,
                               float* __restrict__ C, int ldc, int c_col0,
                               int M, int K, int act)
{
  __shared__ __align__(16) unsigned short sA[BM * LDA_S];
  __shared__ __align__(16) unsigned short sB[BN * LDB_S];

  const int tid  = threadIdx.x;
  const int m0   = blockIdx.x * BM;
  const int n0   = blockIdx.y * BN;
  const int wid  = tid >> 5;
  const int lane = tid & 31;
  const int wm   = wid & 3;    // 0..3  -> m offset wm*32
  const int wn   = wid >> 2;   // 0..1  -> n offset wn*32

  v8f acc[2][2] = {};

  for (int k0 = 0; k0 < K; k0 += BK) {
    __syncthreads();
    // --- load A tile: BM x BK fp32 -> bf16 LDS (1024 float4, 4 per thread) ---
#pragma unroll
    for (int i = 0; i < 4; ++i) {
      int idx = tid + i * 256;
      int r   = idx >> 3;          // 8 float4 per row
      int c4  = idx & 7;
      float4 vv = make_float4(0.f, 0.f, 0.f, 0.f);
      int gr = m0 + r;
      if (gr < M) vv = *(const float4*)(A + (size_t)gr * lda + k0 + c4 * 4);
      unsigned short* p = &sA[r * LDA_S + c4 * 4];
      p[0] = f2bf(vv.x); p[1] = f2bf(vv.y); p[2] = f2bf(vv.z); p[3] = f2bf(vv.w);
    }
    // --- load W tile: BK x BN fp32, store transposed [n][k] (512 float4) ---
#pragma unroll
    for (int i = 0; i < 2; ++i) {
      int idx = tid + i * 256;
      int kr  = idx >> 4;          // 16 float4 per row of 64 cols
      int c4  = idx & 15;
      float4 vv = *(const float4*)(W + (size_t)(k0 + kr) * ldw + n0 + c4 * 4);
      sB[(c4 * 4 + 0) * LDB_S + kr] = f2bf(vv.x);
      sB[(c4 * 4 + 1) * LDB_S + kr] = f2bf(vv.y);
      sB[(c4 * 4 + 2) * LDB_S + kr] = f2bf(vv.z);
      sB[(c4 * 4 + 3) * LDB_S + kr] = f2bf(vv.w);
    }
    __syncthreads();

    // --- build fragments per the 16-bit A/B VGPR layout ---
    const int kb   = (lane < 16) ? 0 : 8;   // K half-offset per lane group
    const int rsel = lane & 15;
    Frag16 aF[2], bF[2];
#pragma unroll
    for (int mi = 0; mi < 2; ++mi) {
      const unsigned short* p = &sA[(wm * 32 + mi * 16 + rsel) * LDA_S + kb];
      aF[mi].q[0] = *(const uint4*)(p);
      aF[mi].q[1] = *(const uint4*)(p + 16);
    }
#pragma unroll
    for (int ni = 0; ni < 2; ++ni) {
      const unsigned short* p = &sB[(wn * 32 + ni * 16 + rsel) * LDB_S + kb];
      bF[ni].q[0] = *(const uint4*)(p);
      bF[ni].q[1] = *(const uint4*)(p + 16);
    }
#pragma unroll
    for (int mi = 0; mi < 2; ++mi)
#pragma unroll
      for (int ni = 0; ni < 2; ++ni)
        acc[mi][ni] = __builtin_amdgcn_wmma_f32_16x16x32_bf16(
            false, aF[mi].v, false, bF[ni].v, (short)0, acc[mi][ni], false, false);
  }

  // --- epilogue: bias, optional residual addend, optional relu, store ---
  const int rsel  = lane & 15;
  const int rhalf = (lane < 16) ? 0 : 8;
#pragma unroll
  for (int mi = 0; mi < 2; ++mi) {
#pragma unroll
    for (int ni = 0; ni < 2; ++ni) {
      int gcol = n0 + wn * 32 + ni * 16 + rsel;        // output feature index
      float bv = bias ? bias[gcol] : 0.f;
#pragma unroll
      for (int r = 0; r < 8; ++r) {
        int grow = m0 + wm * 32 + mi * 16 + rhalf + r;
        if (grow < M) {
          float val = acc[mi][ni][r] + bv;
          if (addend) val += addend[(size_t)grow * ld_add + gcol];
          if (act) val = fmaxf(val, 0.f);
          C[(size_t)grow * ldc + c_col0 + gcol] = val;
        }
      }
    }
  }
}

// ---------------------------------------------------------------------------
// Elementwise / fill kernels
// ---------------------------------------------------------------------------
__global__ void fill_f32(float* __restrict__ p, float v, int n) {
  int i = blockIdx.x * blockDim.x + threadIdx.x;
  if (i < n) p[i] = v;
}

__global__ void relu_inplace(float* __restrict__ p, int n) {
  int i = blockIdx.x * blockDim.x + threadIdx.x;
  if (i < n) p[i] = fmaxf(p[i], 0.f);
}

// h = (1 + eps) * x + agg   (in-place into agg)
__global__ void gin_combine(const float* __restrict__ x, float* __restrict__ h,
                            const float* __restrict__ eps, int n) {
  int i = blockIdx.x * blockDim.x + threadIdx.x;
  if (i >= n) return;
  float e = 1.f + *eps;
  h[i] = e * x[i] + h[i];
}

// ---------------------------------------------------------------------------
// SpMM scatter: agg[row[e]] += vals[e] * X[col[e]]   (float4 per thread chunk)
// ---------------------------------------------------------------------------
__global__ void spmm_scatter(const int* __restrict__ rowi, const int* __restrict__ coli,
                             const float* __restrict__ vals, const float* __restrict__ X,
                             float* __restrict__ agg, int K, int E) {
  int K4 = K >> 2;
  long i = (long)blockIdx.x * blockDim.x + threadIdx.x;
  if (i >= (long)E * K4) return;
  int c4 = (int)(i % K4);
  int e  = (int)(i / K4);
  float val = vals[e];
  const float4 xv = *(const float4*)(X + (size_t)coli[e] * K + c4 * 4);
  float* a = agg + (size_t)rowi[e] * K + c4 * 4;
  atomicAdd(a + 0, val * xv.x);
  atomicAdd(a + 1, val * xv.y);
  atomicAdd(a + 2, val * xv.z);
  atomicAdd(a + 3, val * xv.w);
}

// ---------------------------------------------------------------------------
// LayerNorm(relu(x)) over 64 features, wave per row
// ---------------------------------------------------------------------------
__global__ void relu_ln64(const float* __restrict__ X, const float* __restrict__ g,
                          const float* __restrict__ be, float* __restrict__ Y, int N) {
  int row  = blockIdx.x * (blockDim.x >> 5) + (threadIdx.x >> 5);
  int lane = threadIdx.x & 31;
  if (row >= N) return;
  const float* xp = X + (size_t)row * 64;
  float a  = fmaxf(xp[lane], 0.f);
  float b2 = fmaxf(xp[lane + 32], 0.f);
  float s = a + b2, sq = a * a + b2 * b2;
#pragma unroll
  for (int o = 16; o > 0; o >>= 1) { s += __shfl_xor(s, o, 32); sq += __shfl_xor(sq, o, 32); }
  float mean = s * (1.f / 64.f);
  float var  = sq * (1.f / 64.f) - mean * mean;
  float inv  = rsqrtf(fmaxf(var, 0.f) + 1e-5f);
  float* yp = Y + (size_t)row * 64;
  yp[lane]      = (a  - mean) * inv * g[lane]      + be[lane];
  yp[lane + 32] = (b2 - mean) * inv * g[lane + 32] + be[lane + 32];
}

// ---------------------------------------------------------------------------
// Edge attention: logits, segment max, exp+segment sum, scatter
// ---------------------------------------------------------------------------
__global__ void attn_logits(const float* __restrict__ q, const float* __restrict__ k,
                            const int* __restrict__ src, const int* __restrict__ dst,
                            float* __restrict__ logits, int E, int heads, float scale) {
  int w    = blockIdx.x * (blockDim.x >> 5) + (threadIdx.x >> 5);
  int lane = threadIdx.x & 31;
  if (w >= E) return;
  const float* qp = q + (size_t)dst[w] * heads * 64;
  const float* kp = k + (size_t)src[w] * heads * 64;
  for (int h = 0; h < heads; ++h) {
    float p = qp[h * 64 + lane] * kp[h * 64 + lane]
            + qp[h * 64 + lane + 32] * kp[h * 64 + lane + 32];
#pragma unroll
    for (int o = 16; o > 0; o >>= 1) p += __shfl_xor(p, o, 32);
    if (lane == 0) logits[(size_t)w * heads + h] = p * scale;
  }
}

__device__ inline void atomicMaxF(float* a, float v) {
  if (v >= 0.f) atomicMax((int*)a, __float_as_int(v));
  else          atomicMin((unsigned int*)a, __float_as_uint(v));
}

__global__ void seg_max(const float* __restrict__ logits, const int* __restrict__ dst,
                        float* __restrict__ m, int E, int heads) {
  int i = blockIdx.x * blockDim.x + threadIdx.x;
  if (i >= E * heads) return;
  int e = i / heads, h = i - e * heads;
  atomicMaxF(&m[(size_t)dst[e] * heads + h], logits[i]);
}

__global__ void exp_sum(float* __restrict__ logits, const int* __restrict__ dst,
                        const float* __restrict__ m, float* __restrict__ z,
                        int E, int heads) {
  int i = blockIdx.x * blockDim.x + threadIdx.x;
  if (i >= E * heads) return;
  int e = i / heads, h = i - e * heads;
  float ex = __expf(logits[i] - m[(size_t)dst[e] * heads + h]);
  logits[i] = ex;
  atomicAdd(&z[(size_t)dst[e] * heads + h], ex);
}

__global__ void attn_scatter(const float* __restrict__ ew, const float* __restrict__ z,
                             const float* __restrict__ v, const int* __restrict__ src,
                             const int* __restrict__ dst, float* __restrict__ out,
                             int ldo, int col0, int E, int heads) {
  long i = (long)blockIdx.x * blockDim.x + threadIdx.x;
  long tot = (long)E * heads * 16;
  if (i >= tot) return;
  int qd = (int)(i & 15);
  long t = i >> 4;
  int h  = (int)(t % heads);
  int e  = (int)(t / heads);
  int s  = src[e], d = dst[e];
  float alpha = ew[(size_t)e * heads + h] / z[(size_t)d * heads + h];
  const float4 vv = *(const float4*)(v + (size_t)s * heads * 64 + h * 64 + qd * 4);
  float* o = out + (size_t)d * ldo + col0 + h * 64 + qd * 4;
  atomicAdd(o + 0, alpha * vv.x);
  atomicAdd(o + 1, alpha * vv.y);
  atomicAdd(o + 2, alpha * vv.z);
  atomicAdd(o + 3, alpha * vv.w);
}

// ---------------------------------------------------------------------------
// Host orchestration
// ---------------------------------------------------------------------------
static const int Nn = 30000, Ee = 320000, Fd = 256;

static inline void gemm(hipStream_t st, const float* A, int lda, const float* W, int Mo,
                        const float* bias, const float* addend, int ld_add,
                        float* C, int ldc, int c_col0, int M, int K, int act) {
  dim3 g((M + BM - 1) / BM, Mo / BN);
  gemm_bf16_wmma<<<g, 256, 0, st>>>(A, lda, W, Mo, bias, addend, ld_add,
                                    C, ldc, c_col0, M, K, act);
}

static inline void fill(hipStream_t st, float* p, float v, int n) {
  fill_f32<<<(n + 255) / 256, 256, 0, st>>>(p, v, n);
}

static void run_graph(void* const* din, int ebase, int gbase, int tbase, int fbase,
                      float* outp,
                      float* bufA, float* bufB, float* bufC, float* bufD,
                      float* t0, float* t1b, float* t2b, float* fused,
                      float* logits, float* mbuf, float* zbuf, hipStream_t st)
{
  const int*   eidx = (const int*)din[ebase];
  const float* vals = (const float*)din[ebase + 1];
  const float* x    = (const float*)din[ebase + 2];
  const int* erow = eidx;          // edge_index[0]
  const int* ecol = eidx + Ee;     // edge_index[1]
  auto P = [&](int i) { return (const float*)din[i]; };
  const float NEG_INF = -__builtin_huge_valf();

  // ======================= GIN branch =======================
  fill(st, bufA, 0.f, Nn * Fd);
  { long nt = (long)Ee * (Fd >> 2);
    spmm_scatter<<<(unsigned)((nt + 255) / 256), 256, 0, st>>>(erow, ecol, vals, x, bufA, Fd, Ee); }
  gin_combine<<<(Nn * Fd + 255) / 256, 256, 0, st>>>(x, bufA, P(gbase + 0), Nn * Fd);
  gemm(st, bufA, Fd, P(gbase + 2),  64, P(gbase + 3),  nullptr, 0, t0,  64, 0, Nn, Fd, 1); // relu(h@W1a+b1a)
  gemm(st, t0,   64, P(gbase + 4),  64, P(gbase + 5),  nullptr, 0, t1b, 64, 0, Nn, 64, 0); // @W1b+b1b
  relu_ln64<<<(Nn + 7) / 8, 256, 0, st>>>(t1b, P(gbase + 12), P(gbase + 13), t2b, Nn);     // hn

  fill(st, t0, 0.f, Nn * 64);
  { long nt = (long)Ee * 16;
    spmm_scatter<<<(unsigned)((nt + 255) / 256), 256, 0, st>>>(erow, ecol, vals, t2b, t0, 64, Ee); }
  gin_combine<<<(Nn * 64 + 255) / 256, 256, 0, st>>>(t2b, t0, P(gbase + 1), Nn * 64);
  gemm(st, t0,  64, P(gbase + 6),  64, P(gbase + 7),  nullptr, 0, t1b, 64, 0, Nn, 64, 1);
  gemm(st, t1b, 64, P(gbase + 8),  64, P(gbase + 9),  nullptr, 0, t0,  64, 0, Nn, 64, 0);
  relu_ln64<<<(Nn + 7) / 8, 256, 0, st>>>(t0, P(gbase + 14), P(gbase + 15), t1b, Nn);      // g2n
  // gin_out = g2n + x@Wr + br  -> fused cols [0,64)
  gemm(st, x, Fd, P(gbase + 10), 64, P(gbase + 11), t1b, 64, fused, 128, 0, Nn, Fd, 0);

  // ================== Transformer layer 1 (heads=4) ==================
  gemm(st, x, Fd, P(tbase + 0), 256, P(tbase + 1), nullptr, 0, bufA, 256, 0, Nn, Fd, 0); // q
  gemm(st, x, Fd, P(tbase + 2), 256, P(tbase + 3), nullptr, 0, bufB, 256, 0, Nn, Fd, 0); // k
  gemm(st, x, Fd, P(tbase + 4), 256, P(tbase + 5), nullptr, 0, bufC, 256, 0, Nn, Fd, 0); // v
  gemm(st, x, Fd, P(tbase + 6), 256, P(tbase + 7), nullptr, 0, bufD, 256, 0, Nn, Fd, 0); // skip -> out

  fill(st, mbuf, NEG_INF, Nn * 4);
  fill(st, zbuf, 0.f, Nn * 4);
  attn_logits<<<(Ee + 7) / 8, 256, 0, st>>>(bufA, bufB, erow, ecol, logits, Ee, 4, 0.125f);
  seg_max<<<(Ee * 4 + 255) / 256, 256, 0, st>>>(logits, ecol, mbuf, Ee, 4);
  exp_sum<<<(Ee * 4 + 255) / 256, 256, 0, st>>>(logits, ecol, mbuf, zbuf, Ee, 4);
  { long nt = (long)Ee * 4 * 16;
    attn_scatter<<<(unsigned)((nt + 255) / 256), 256, 0, st>>>(logits, zbuf, bufC, erow, ecol, bufD, 256, 0, Ee, 4); }
  relu_inplace<<<(Nn * 256 + 255) / 256, 256, 0, st>>>(bufD, Nn * 256);

  // ================== Transformer layer 2 (heads=1) ==================
  gemm(st, bufD, 256, P(tbase + 8),  64, P(tbase + 9),  nullptr, 0, t0,  64, 0, Nn, 256, 0); // q2
  gemm(st, bufD, 256, P(tbase + 10), 64, P(tbase + 11), nullptr, 0, t1b, 64, 0, Nn, 256, 0); // k2
  gemm(st, bufD, 256, P(tbase + 12), 64, P(tbase + 13), nullptr, 0, t2b, 64, 0, Nn, 256, 0); // v2
  gemm(st, bufD, 256, P(tbase + 14), 64, P(tbase + 15), nullptr, 0, fused, 128, 64, Nn, 256, 0); // s2 -> fused cols [64,128)

  fill(st, mbuf, NEG_INF, Nn);
  fill(st, zbuf, 0.f, Nn);
  attn_logits<<<(Ee + 7) / 8, 256, 0, st>>>(t0, t1b, erow, ecol, logits, Ee, 1, 0.125f);
  seg_max<<<(Ee + 255) / 256, 256, 0, st>>>(logits, ecol, mbuf, Ee, 1);
  exp_sum<<<(Ee + 255) / 256, 256, 0, st>>>(logits, ecol, mbuf, zbuf, Ee, 1);
  { long nt = (long)Ee * 16;
    attn_scatter<<<(unsigned)((nt + 255) / 256), 256, 0, st>>>(logits, zbuf, t2b, erow, ecol, fused, 128, 64, Ee, 1); }

  // ================== Fusion ==================
  gemm(st, fused, 128, P(fbase + 0), 64, P(fbase + 1), nullptr, 0, outp, 64, 0, Nn, 128, 1);
}

extern "C" void kernel_launch(void* const* d_in, const int* in_sizes, int n_in,
                              void* d_out, int out_size, void* d_ws, size_t ws_size,
                              hipStream_t stream) {
  (void)in_sizes; (void)n_in; (void)out_size; (void)ws_size;
  // Workspace carve (aligned to 256B)
  size_t off = 0;
  auto carve = [&](size_t bytes) {
    void* p = (char*)d_ws + off;
    off += (bytes + 255) & ~(size_t)255;
    return (float*)p;
  };
  float* bufA   = carve((size_t)Nn * 256 * 4);
  float* bufB   = carve((size_t)Nn * 256 * 4);
  float* bufC   = carve((size_t)Nn * 256 * 4);
  float* bufD   = carve((size_t)Nn * 256 * 4);
  float* t0     = carve((size_t)Nn * 64 * 4);
  float* t1b    = carve((size_t)Nn * 64 * 4);
  float* t2b    = carve((size_t)Nn * 64 * 4);
  float* fused  = carve((size_t)Nn * 128 * 4);
  float* logits = carve((size_t)Ee * 4 * 4);
  float* mbuf   = carve((size_t)Nn * 4 * 4);
  float* zbuf   = carve((size_t)Nn * 4 * 4);

  float* out = (float*)d_out;
  // Input layout (flattened setup_inputs dict order):
  //  0 drug_edge_index, 1 drug_adj_values, 2 drug_sim_feat,
  //  3 dis_edge_index,  4 dis_adj_values,  5 disease_sim_feat,
  //  6..21  gin_drug {eps1,eps2,W1a,b1a,W1b,b1b,W2a,b2a,W2b,b2b,Wr,br,g1,be1,g2,be2}
  // 22..37  gin_dis
  // 38..53  tr_drug {t1:Wq,bq,Wk,bk,Wv,bv,Ws,bs; t2: same}
  // 54..69  tr_dis
  // 70..71  fus_drug {W,b}; 72..73 fus_dis {W,b}
  run_graph(d_in, 0, 6, 38, 70, out,
            bufA, bufB, bufC, bufD, t0, t1b, t2b, fused, logits, mbuf, zbuf, stream);
  run_graph(d_in, 3, 22, 54, 72, out + (size_t)Nn * 64,
            bufA, bufB, bufC, bufD, t0, t1b, t2b, fused, logits, mbuf, zbuf, stream);
}
